// MultiHeadSelfAttention_22204980920452
// MI455X (gfx1250) — matrix operand order repaired and checked by reading, not compiled, not run on hardware
//
#include <hip/hip_runtime.h>
#include <hip/hip_bf16.h>
#include <type_traits>

typedef __attribute__((ext_vector_type(16))) _Float16 v16h;
typedef __attribute__((ext_vector_type(8)))  _Float16 v8h;
typedef __attribute__((ext_vector_type(8)))  float    v8f;

namespace {
constexpr int kBatch  = 2;
constexpr int kHeads  = 16;
constexpr int kSeq    = 2048;
constexpr int kDModel = 1024;
constexpr int kDHead  = 64;
constexpr int kQTile  = 64;   // query rows per block (4 waves x 16)
constexpr int kKTile  = 64;   // keys per main-loop iteration
constexpr int kStr    = 72;   // padded LDS row stride (halves), 144 B (16B-aligned)
constexpr float kRopeLn = 0.2878231366242f;             // ln(10000)/32
constexpr float kQScale = 0.18033688011112042f;         // 0.125 * log2(e)
constexpr size_t kHalfPerTensor = (size_t)kBatch * kHeads * kSeq * kDHead; // 4,194,304
// workspace layout: [qh | kh | vt], 8 MB each, 24 MB total
}

union V16 { v16h v; v8h h[2]; };

static __device__ inline v8f wmma16x16x32(v16h a, v16h b, v8f c) {
  return __builtin_amdgcn_wmma_f32_16x16x32_f16(false, a, false, b, (short)0, c,
                                                false, false);
}

// XOR lane swap via ds_swizzle (group-of-32 mode, and_mask=0x1f, imm pattern).
template <int MASK>
static __device__ __forceinline__ float swz_xor(float x) {
  const int imm = (MASK << 10) | 0x1f;
  return __int_as_float(__builtin_amdgcn_ds_swizzle(__float_as_int(x), imm));
}

// Async copy of one 16-byte chunk global -> LDS (CDNA5, ASYNCcnt-tracked).
static __device__ __forceinline__ void async_copy_b128(unsigned lds_off,
                                                       const _Float16* g) {
  unsigned long long ga = (unsigned long long)(uintptr_t)g;
  asm volatile("global_load_async_to_lds_b128 %0, %1, off"
               :: "v"(lds_off), "v"(ga) : "memory");
}

// ---------------------------------------------------------------------------
// Kernel 1: one-shot RoPE + f16 conversion.
//   qh : RoPE(Q) * 0.125*log2(e)  (scores come out in log2 domain)
//   kh : RoPE(K), row-major f16
//   vt : V transposed [b,h][dim][seq], f16
// ---------------------------------------------------------------------------
__global__ __launch_bounds__(256) void rope_prep_kernel(
    const float* __restrict__ q, const float* __restrict__ k,
    const float* __restrict__ v, _Float16* __restrict__ qh,
    _Float16* __restrict__ kh, _Float16* __restrict__ vt) {
  const size_t idx = (size_t)blockIdx.x * 256 + threadIdx.x;  // one dim-pair
  const int pairsPerRow = kDModel / 2;                        // 512
  const int p = (int)(idx % pairsPerRow);
  const int s = (int)((idx / pairsPerRow) % kSeq);
  const int b = (int)(idx / ((size_t)pairsPerRow * kSeq));
  const int h = p >> 5;                                       // 32 pairs/head
  const int i = p & 31;                                       // pair in head

  const size_t srcOff  = ((size_t)b * kSeq + s) * kDModel + h * kDHead + 2 * i;
  const size_t headIdx = (size_t)b * kHeads + h;
  const size_t dstOff  = (headIdx * kSeq + s) * kDHead + 2 * i;

  const float inv = __expf(-kRopeLn * (float)i);
  const float ang = (float)s * inv;
  float sn, cs; __sincosf(ang, &sn, &cs);

  {
    const float xe = q[srcOff], xo = q[srcOff + 1];
    qh[dstOff]     = (_Float16)(kQScale * (xe * cs - xo * sn));
    qh[dstOff + 1] = (_Float16)(kQScale * (xe * sn + xo * cs));
  }
  {
    const float xe = k[srcOff], xo = k[srcOff + 1];
    kh[dstOff]     = (_Float16)(xe * cs - xo * sn);
    kh[dstOff + 1] = (_Float16)(xe * sn + xo * cs);
  }
  {
    const size_t vtBase = headIdx * kDHead;
    vt[(vtBase + 2 * i)     * kSeq + s] = (_Float16)v[srcOff];
    vt[(vtBase + 2 * i + 1) * kSeq + s] = (_Float16)v[srcOff + 1];
  }
}

// ---------------------------------------------------------------------------
// Kernel 2: flash attention over prepared f16 tensors (64-key tiles).
// ---------------------------------------------------------------------------
__global__ __launch_bounds__(128) void fa_kernel(const _Float16* __restrict__ qh,
                                                 const _Float16* __restrict__ kh,
                                                 const _Float16* __restrict__ vt,
                                                 float* __restrict__ out) {
  __shared__ __align__(16) _Float16 Kb[2][kKTile * kStr];   // 2 x 9216 B
  __shared__ __align__(16) _Float16 Vb[2][kDHead * kStr];   // 2 x 9216 B
  __shared__ __align__(16) _Float16 Ps[4][16 * kStr];       // 4 x 2304 B

  const int bid   = blockIdx.x;
  const int nqblk = kSeq / kQTile;                 // 32
  const int qblk  = bid % nqblk;
  const int head  = (bid / nqblk) % kHeads;
  const int batch = bid / (nqblk * kHeads);

  const int tid  = threadIdx.x;
  const int lane = tid & 31;
  const int wave = tid >> 5;

  const size_t headIdx = (size_t)batch * kHeads + head;
  const _Float16* qhh = qh + headIdx * kSeq * kDHead;
  const _Float16* khh = kh + headIdx * kSeq * kDHead;
  const _Float16* vth = vt + headIdx * kDHead * kSeq;
  float* og = out + (size_t)batch * kSeq * kDModel + (size_t)head * kDHead;

  const int Q0    = qblk * kQTile;
  const int qbase = Q0 + wave * 16;

  // Stage one 64-key tile (K rows + V^T rows) into LDS buffer `buf`:
  // 128 threads x 8 async chunks x 16 B = 16 KB.
  auto stage_tile = [&](int buf, int K0) {
    const _Float16* kg = khh + (size_t)K0 * kDHead;
#pragma unroll
    for (int c = 0; c < 4; ++c) {
      const int ch  = tid * 4 + c;            // 0..511
      const int key = ch >> 3;                // 64 keys x 8 chunks
      const int sub = ch & 7;
      const unsigned loff =
          (unsigned)(uintptr_t)&Kb[buf][key * kStr + sub * 8];
      async_copy_b128(loff, kg + key * kDHead + sub * 8);
    }
    const _Float16* vg = vth + K0;
#pragma unroll
    for (int c = 0; c < 4; ++c) {
      const int ch  = tid * 4 + c;            // 0..511
      const int d   = ch >> 3;                // 64 dims x 8 chunks
      const int sub = ch & 7;
      const unsigned loff =
          (unsigned)(uintptr_t)&Vb[buf][d * kStr + sub * 8];
      async_copy_b128(loff, vg + (size_t)d * kSeq + sub * 8);
    }
  };

  // ---- per-wave Q A-fragments straight from global ----
  const int ar = lane & 15;
  const int ab = (lane >> 4) * 16;
  V16 qa0, qa1;
  {
    const _Float16* qr = qhh + (size_t)(qbase + ar) * kDHead;
    qa0.h[0] = *(const v8h*)(qr + ab);
    qa0.h[1] = *(const v8h*)(qr + ab + 8);
    qa1.h[0] = *(const v8h*)(qr + 32 + ab);
    qa1.h[1] = *(const v8h*)(qr + 32 + ab + 8);
  }

  v16h ones;
#pragma unroll
  for (int i = 0; i < 16; ++i) ones[i] = (_Float16)1.0f;

  v8f o0 = {}, o1 = {}, o2 = {}, o3 = {};
  float mrow[8], lrow[8];
#pragma unroll
  for (int r = 0; r < 8; ++r) { mrow[r] = -1e30f; lrow[r] = 0.f; }

  const int bn   = lane & 15;
  const int bk   = (lane >> 4) * 16;
  const int prow = (lane >> 4) * 8;

  // Process one staged 64-key tile. MASK is compile-time: only the diagonal
  // tile (K0 == Q0) carries the causal-mask compares.
  auto process_tile = [&](int cur, int K0, auto maskTag) {
    constexpr bool MASK = decltype(maskTag)::value;

    // ---- scores: S[16x64] = Q(16x64) * K^T, four 16x16 C tiles ----
    v8f sc[4];
#pragma unroll
    for (int t = 0; t < 4; ++t) {
      const _Float16* kp = &Kb[cur][(t * 16 + bn) * kStr];
      V16 b0, b1;
      b0.h[0] = *(const v8h*)(kp + bk);      b0.h[1] = *(const v8h*)(kp + bk + 8);
      b1.h[0] = *(const v8h*)(kp + 32 + bk); b1.h[1] = *(const v8h*)(kp + 32 + bk + 8);
      v8f s = {};
      s = wmma16x16x32(qa0.v, b0.v, s);
      s = wmma16x16x32(qa1.v, b1.v, s);
      sc[t] = s;
    }

    // ---- online softmax (log2 domain); row max via ds_swizzle ----
    _Float16* pw = &Ps[wave][0];
    float alpha_s[8];
#pragma unroll
    for (int r = 0; r < 8; ++r) {
      const int row = qbase + prow + r;
      float e[4], mx = -1e30f;
#pragma unroll
      for (int t = 0; t < 4; ++t) {
        float x = sc[t][r];
        if (MASK && (K0 + t * 16 + bn > row)) x = -1e30f;
        e[t] = x;
        mx = fmaxf(mx, x);
      }
      mx = fmaxf(mx, swz_xor<1>(mx));
      mx = fmaxf(mx, swz_xor<2>(mx));
      mx = fmaxf(mx, swz_xor<4>(mx));
      mx = fmaxf(mx, swz_xor<8>(mx));
      const float m2 = fmaxf(mrow[r], mx);
      alpha_s[r] = exp2f(mrow[r] - m2);
      mrow[r] = m2;
#pragma unroll
      for (int t = 0; t < 4; ++t) {
        e[t] = exp2f(e[t] - m2);
        pw[(prow + r) * kStr + t * 16 + bn] = (_Float16)e[t];
      }
      o0[r] *= alpha_s[r]; o1[r] *= alpha_s[r];
      o2[r] *= alpha_s[r]; o3[r] *= alpha_s[r];
    }

    // ---- reload P as two A fragments (K = 0..31, 32..63) ----
    asm volatile("s_wait_dscnt 0x0" ::: "memory");
    V16 pa0, pa1;
    {
      const _Float16* pr = &pw[ar * kStr];
      pa0.h[0] = *(const v8h*)(pr + ab);
      pa0.h[1] = *(const v8h*)(pr + ab + 8);
      pa1.h[0] = *(const v8h*)(pr + 32 + ab);
      pa1.h[1] = *(const v8h*)(pr + 32 + ab + 8);
    }

    // ---- row sums via WMMA: rsC = P * ones(64x16); every column = rowsum ----
    v8f rsC = {};
    rsC = wmma16x16x32(pa0.v, ones, rsC);
    rsC = wmma16x16x32(pa1.v, ones, rsC);
#pragma unroll
    for (int r = 0; r < 8; ++r) lrow[r] = lrow[r] * alpha_s[r] + rsC[r];

    // ---- O += P(16x64) * V(64x64), four output tiles x two K chunks ----
#pragma unroll
    for (int d = 0; d < 4; ++d) {
      const _Float16* vp = &Vb[cur][(d * 16 + bn) * kStr];
      V16 vb0, vb1;
      vb0.h[0] = *(const v8h*)(vp + bk);      vb0.h[1] = *(const v8h*)(vp + bk + 8);
      vb1.h[0] = *(const v8h*)(vp + 32 + bk); vb1.h[1] = *(const v8h*)(vp + 32 + bk + 8);
      v8f& acc = (d == 0) ? o0 : (d == 1) ? o1 : (d == 2) ? o2 : o3;
      acc = wmma16x16x32(pa0.v, vb0.v, acc);
      acc = wmma16x16x32(pa1.v, vb1.v, acc);
    }
  };

  stage_tile(0, 0);                      // prologue prefetch

  // ---- clean tiles: all 64 keys < every query row of this block ----
  for (int kt = 0; kt < qblk; ++kt) {
    const int cur = kt & 1;
    asm volatile("s_wait_asynccnt 0x0" ::: "memory"); // my copies landed
    __syncthreads();                                  // everyone's copies landed
    stage_tile(1 - cur, (kt + 1) * kKTile);           // overlap next tile
    process_tile(cur, kt * kKTile, std::false_type{});
  }

  // ---- diagonal tile (K0 == Q0): causal mask active ----
  {
    const int cur = qblk & 1;
    asm volatile("s_wait_asynccnt 0x0" ::: "memory");
    __syncthreads();
    process_tile(cur, Q0, std::true_type{});
  }

  // ---- epilogue: normalize rows and store fp32 ----
#pragma unroll
  for (int r = 0; r < 8; ++r) {
    const float inv = 1.0f / lrow[r];
    const int row = qbase + prow + r;
    float* orow = og + (size_t)row * kDModel;
    orow[ 0 + bn] = o0[r] * inv;
    orow[16 + bn] = o1[r] * inv;
    orow[32 + bn] = o2[r] * inv;
    orow[48 + bn] = o3[r] * inv;
  }
}

extern "C" void kernel_launch(void* const* d_in, const int* in_sizes, int n_in,
                              void* d_out, int out_size, void* d_ws, size_t ws_size,
                              hipStream_t stream) {
  (void)in_sizes; (void)n_in; (void)out_size; (void)ws_size; // needs 24 MB ws
  const float* q = (const float*)d_in[0];
  const float* k = (const float*)d_in[1];
  const float* v = (const float*)d_in[2];
  float* out = (float*)d_out;

  _Float16* qh = (_Float16*)d_ws;
  _Float16* kh = qh + kHalfPerTensor;
  _Float16* vt = kh + kHalfPerTensor;

  const int prepGrid = (int)((size_t)kBatch * kSeq * (kDModel / 2) / 256); // 8192
  rope_prep_kernel<<<prepGrid, 256, 0, stream>>>(q, k, v, qh, kh, vt);

  const int faGrid = kBatch * kHeads * (kSeq / kQTile);  // 1024
  fa_kernel<<<faGrid, 128, 0, stream>>>(qh, kh, vt, out);
}
